// FasterConv2d_19301583028401
// MI455X (gfx1250) — matrix operand order repaired and checked by reading, not compiled, not run on hardware
//
#include <hip/hip_runtime.h>
#include <hip/hip_bf16.h>

typedef __bf16 bf16;
typedef bf16  v16bf __attribute__((ext_vector_type(16)));
typedef bf16  v8bf  __attribute__((ext_vector_type(8)));
typedef float v8f   __attribute__((ext_vector_type(8)));
typedef unsigned int u32x4 __attribute__((ext_vector_type(4)));
typedef int          i32x8 __attribute__((ext_vector_type(8)));
typedef int          i32x4 __attribute__((ext_vector_type(4)));

#define BLOCK_M 128
#define BLOCK_N 128
#define KC       32
#define LDSS     40   // KC + 8 bf16 pad: row stride 80B = 5 x 16B -> conflict-spread b128 reads

#define WT_PLANE (9 * 256 * 128)                     // elements per hi/lo plane
#define WT_BYTES (2u * WT_PLANE * sizeof(bf16))      // 1,179,648 B workspace need

// Split fp32 into hi/lo bf16 (both RNE) for bf16x3 emulated-fp32 matmul.
__device__ __forceinline__ void split_bf16(float f, bf16& hi, bf16& lo) {
  unsigned u  = __builtin_bit_cast(unsigned, f);
  unsigned rh = u + 0x7FFFu + ((u >> 16) & 1u);
  unsigned short hs = (unsigned short)(rh >> 16);
  float hf = __builtin_bit_cast(float, (unsigned)hs << 16);
  float l  = f - hf;
  unsigned ul = __builtin_bit_cast(unsigned, l);
  unsigned rl = ul + 0x7FFFu + ((ul >> 16) & 1u);
  hi = __builtin_bit_cast(bf16, hs);
  lo = __builtin_bit_cast(bf16, (unsigned short)(rl >> 16));
}

// Issue a 2D TDM tile load: 128 lines x 32 bf16, line stride 128 elems, with
// hardware LDS padding 16B per 64B row -> LDSS(=40)-element LDS row stride.
__device__ __forceinline__ void tdm_load_tile(unsigned lds_byte_off,
                                              unsigned long long gaddr) {
  u32x4 g0;
  g0[0] = 1u;                                  // count=1, user descriptor
  g0[1] = lds_byte_off;                        // lds_addr
  g0[2] = (unsigned)(gaddr & 0xFFFFFFFFull);   // global_addr[31:0]
  g0[3] = (unsigned)(gaddr >> 32) | (2u << 30);// global_addr[56:32] | type=2

  i32x8 g1;
  g1[0] = (int)((1u << 16)        // data_size = 2 bytes
              | (1u << 20)        // pad_enable
              | (3u << 22)        // pad_interval: 16 DWORDs (one 64B row)
              | (3u << 25));      // pad_amount:   4 DWORDs (16B)
  g1[1] = (int)(32u << 16);       // tensor_dim0 = 32 (low 16 into d1[31:16])
  g1[2] = (int)(128u << 16);      // tensor_dim1 = 128 (low 16 into d2[31:16])
  g1[3] = (int)(32u << 16);       // tile_dim0 = 32
  g1[4] = (int)128u;              // tile_dim1 = 128, tile_dim2 = 0 (2D)
  g1[5] = (int)128u;              // tensor_dim0_stride = 128 elems
  g1[6] = 0;                      // dim0_stride hi / dim1_stride lo
  g1[7] = 0;

  i32x4 gz; gz[0] = 0; gz[1] = 0; gz[2] = 0; gz[3] = 0;

#if __has_include(<hip/amd_detail/amd_gfx1250_TDM.h>)
  i32x8 gz8; gz8[0]=0; gz8[1]=0; gz8[2]=0; gz8[3]=0; gz8[4]=0; gz8[5]=0; gz8[6]=0; gz8[7]=0;
  __builtin_amdgcn_tensor_load_to_lds(g0, g1, gz, gz, gz8, 0);   // therock 6-arg form
#else
  __builtin_amdgcn_tensor_load_to_lds(g0, g1, gz, gz, 0);        // ROCm 7.2 5-arg form
#endif
}

// One-shot filter repack: OIHW f32 -> [pos][o][i] bf16 hi/lo planes (1.2MB, L2-resident).
__global__ __launch_bounds__(256)
void FilterPrepack(const float* __restrict__ w, bf16* __restrict__ hi,
                   bf16* __restrict__ lo) {
  int i = blockIdx.x * 256 + threadIdx.x;      // < 9*256*128
  int pos = i >> 15;                            // / (256*128)
  int rem = i & 32767;
  int o = rem >> 7, c = rem & 127;
  float v = w[o * 1152 + c * 9 + pos];
  bf16 h, l; split_bf16(v, h, l);
  hi[i] = h; lo[i] = l;
}

__global__ __launch_bounds__(256)
void FasterConv2d_wmma_bf16x3(const float* __restrict__ x,
                              const float* __restrict__ w,
                              const float* __restrict__ bias,
                              float* __restrict__ out,
                              const bf16* __restrict__ wtHi,
                              const bf16* __restrict__ wtLo) {
  // LDS staging tiles: A (pixels x channels) and B^T (outch x channels), hi+lo parts.
  __shared__ __align__(16) bf16 sAh[BLOCK_M * LDSS];
  __shared__ __align__(16) bf16 sAl[BLOCK_M * LDSS];
  __shared__ __align__(16) bf16 sBh[BLOCK_N * LDSS];
  __shared__ __align__(16) bf16 sBl[BLOCK_N * LDSS];

  const int tid   = threadIdx.x;
  const int tileM = blockIdx.x % 784;        // 100352 / 128
  const int tileN = blockIdx.x / 784;        // 256 / 128
  const int m0    = tileM * BLOCK_M;
  const int n0    = tileN * BLOCK_N;

  const int wave  = tid >> 5;
  const int lane  = tid & 31;
  const int g     = lane >> 4;               // lane half (WMMA layout)
  const int ln    = lane & 15;
  const int wm0   = (wave & 3) * 32;         // 4 waves tile M
  const int wn0   = (wave >> 2) * 64;        // 2 waves tile N

  const bool useTdm = (wtHi != nullptr);
  // LDS byte offsets of the B tiles (flat LDS addr: low 32 bits are the LDS offset).
  const unsigned sBhOff = (unsigned)(unsigned long long)(&sBh[0]);
  const unsigned sBlOff = (unsigned)(unsigned long long)(&sBl[0]);

  v8f acc[2][4] = {};                        // 32x64 f32 accum per wave

  for (int pos = 0; pos < 9; ++pos) {
    const int r = pos / 3, s = pos % 3;
    for (int c0 = 0; c0 < 128; c0 += KC) {
      // ---- stage B^T tile: sB[n][c] = filters[n0+n][c0+c][r][s] ----
      if (useTdm) {
        if (wave == 0) {
          // element index into [pos][o][i] plane at tile origin (n0, c0)
          const unsigned long long e =
              (unsigned long long)(pos * 32768 + n0 * 128 + c0);
          tdm_load_tile(sBhOff, (unsigned long long)wtHi + e * 2ull);
          tdm_load_tile(sBlOff, (unsigned long long)wtLo + e * 2ull);
        }
      } else {
        for (int t = tid; t < BLOCK_N * KC; t += 256) {
          int nl = t >> 5, cl = t & (KC - 1);
          float wv = w[(n0 + nl) * 1152 + (c0 + cl) * 9 + pos];
          bf16 h, l; split_bf16(wv, h, l);
          sBh[nl * LDSS + cl] = h;
          sBl[nl * LDSS + cl] = l;
        }
      }
      // ---- stage A tile (im2col gather, zero-pad OOB) ----
      for (int t = tid; t < BLOCK_M * KC; t += 256) {
        int ml = t >> 5, cl = t & (KC - 1);
        int m    = m0 + ml;
        int nimg = m / 3136;
        int hw   = m - nimg * 3136;
        int oh   = hw / 56;
        int ow   = hw - oh * 56;
        int ih   = oh + r - 1;
        int iw   = ow + s - 1;
        float v = 0.0f;
        if ((unsigned)ih < 56u && (unsigned)iw < 56u)
          v = x[nimg * 401408 + (c0 + cl) * 3136 + ih * 56 + iw];
        bf16 h, l; split_bf16(v, h, l);
        sAh[ml * LDSS + cl] = h;
        sAl[ml * LDSS + cl] = l;
      }
      if (useTdm) __builtin_amdgcn_s_wait_tensorcnt(0);
      __syncthreads();

      // ---- load WMMA fragments per ISA 7.12.2 layouts ----
      union AB { v16bf v; v8bf h[2]; };
      v16bf ah[2], al[2], bh[4], bl[4];
#pragma unroll
      for (int mf = 0; mf < 2; ++mf) {
        // A 16x32 bf16: lane (g, ln): VGPR0-3 = K[g*8 .. g*8+7], VGPR4-7 = K[16+g*8 ..]
        const int off = (wm0 + mf * 16 + ln) * LDSS + g * 8;
        AB th, tl;
        th.h[0] = *(const v8bf*)(sAh + off);
        th.h[1] = *(const v8bf*)(sAh + off + 16);
        tl.h[0] = *(const v8bf*)(sAl + off);
        tl.h[1] = *(const v8bf*)(sAl + off + 16);
        ah[mf] = th.v; al[mf] = tl.v;
      }
#pragma unroll
      for (int nf = 0; nf < 4; ++nf) {
        // B 32x16 bf16: lane (g, ln): VGPR0-3 = K[g*8 .. g*8+7], VGPR4-7 = K[16+g*8 ..]
        const int off = (wn0 + nf * 16 + ln) * LDSS + g * 8;
        AB th, tl;
        th.h[0] = *(const v8bf*)(sBh + off);
        th.h[1] = *(const v8bf*)(sBh + off + 16);
        tl.h[0] = *(const v8bf*)(sBl + off);
        tl.h[1] = *(const v8bf*)(sBl + off + 16);
        bh[nf] = th.v; bl[nf] = tl.v;
      }

      // ---- bf16x3: acc += a_hi*b_lo + a_lo*b_hi + a_hi*b_hi ----
#pragma unroll
      for (int mf = 0; mf < 2; ++mf) {
#pragma unroll
        for (int nf = 0; nf < 4; ++nf) {
          acc[mf][nf] = __builtin_amdgcn_wmma_f32_16x16x32_bf16(
              false, ah[mf], false, bl[nf], (short)0, acc[mf][nf], false, false);
          acc[mf][nf] = __builtin_amdgcn_wmma_f32_16x16x32_bf16(
              false, al[mf], false, bh[nf], (short)0, acc[mf][nf], false, false);
          acc[mf][nf] = __builtin_amdgcn_wmma_f32_16x16x32_bf16(
              false, ah[mf], false, bh[nf], (short)0, acc[mf][nf], false, false);
        }
      }
      __syncthreads();
    }
  }

  // ---- epilogue: C/D layout VGPR j -> M = j + 8*g, N = ln; add bias, NCHW store ----
#pragma unroll
  for (int nf = 0; nf < 4; ++nf) {
    const int kg = n0 + wn0 + nf * 16 + ln;
    const float bv = bias[kg];
#pragma unroll
    for (int mf = 0; mf < 2; ++mf) {
#pragma unroll
      for (int j = 0; j < 8; ++j) {
        int m    = m0 + wm0 + mf * 16 + g * 8 + j;
        int nimg = m / 3136;
        int hw   = m - nimg * 3136;
        out[nimg * 802816 + kg * 3136 + hw] = acc[mf][nf][j] + bv;
      }
    }
  }
}

extern "C" void kernel_launch(void* const* d_in, const int* in_sizes, int n_in,
                              void* d_out, int out_size, void* d_ws, size_t ws_size,
                              hipStream_t stream) {
  const float* x  = (const float*)d_in[0];
  const float* w  = (const float*)d_in[1];
  const float* b  = (const float*)d_in[2];
  float* out      = (float*)d_out;

  const bool useTdm = (ws_size >= (size_t)WT_BYTES) && (d_ws != nullptr);
  bf16* wtHi = useTdm ? (bf16*)d_ws : nullptr;
  bf16* wtLo = useTdm ? ((bf16*)d_ws) + WT_PLANE : nullptr;

  if (useTdm) {
    dim3 pgrid(WT_PLANE / 256);
    hipLaunchKernelGGL(FilterPrepack, pgrid, dim3(256), 0, stream, w, wtHi, wtLo);
  }
  dim3 grid(784 * 2);   // (100352/128) M-tiles x (256/128) N-tiles
  dim3 block(256);      // 8 wave32s
  hipLaunchKernelGGL(FasterConv2d_wmma_bf16x3, grid, block, 0, stream,
                     x, w, b, out, wtHi, wtLo);
}
